// AdaLN_86371792322640
// MI455X (gfx1250) — matrix operand order repaired and checked by reading, not compiled, not run on hardware
//
#include <hip/hip_runtime.h>

// AdaLN fused kernel for MI455X (gfx1250, wave32, WMMA).
// M = 4*8192 = 32768 rows, c_a = 768, c_s = 384.
// out[m, :] = LN(a[m]) * sigmoid(LN_w(s[m]) @ Wg^T + bg) + LN_w(s[m]) @ Wb^T

#define C_A       768
#define C_S       384
#define M_TOT     32768
#define ROWS      32          // rows per workgroup
#define SN_STRIDE 392         // bf16 elems per s_n row (384 + 8 pad -> bank-friendly)
#define AN_STRIDE 772         // f32 elems per a_n row (768 + 4 pad)
#define LN_EPS    1e-5f

typedef __attribute__((ext_vector_type(16))) __bf16 v16bf;
typedef __attribute__((ext_vector_type(8)))  float  v8f;

union FragAB { v16bf v; uint4 q[2]; };

__device__ __forceinline__ unsigned short f2bf(float f) {
  unsigned int u = __float_as_uint(f);
  u += 0x7fffu + ((u >> 16) & 1u);          // round-to-nearest-even
  return (unsigned short)(u >> 16);
}

// ---------------------------------------------------------------------------
// Prep: fp32 -> bf16 weight conversion into workspace (gamma then beta).
// Needs ws_size >= 2 * 768 * 384 * 2 = 1,179,648 bytes.
// ---------------------------------------------------------------------------
__global__ __launch_bounds__(256) void adaln_prep_weights(
    const float* __restrict__ wg, const float* __restrict__ wb,
    unsigned short* __restrict__ out_bf) {
  const int n4 = (C_A * C_S) / 4;           // float4s per matrix = 73728
  int idx = blockIdx.x * blockDim.x + threadIdx.x;
  if (idx >= 2 * n4) return;
  const float* src = (idx < n4) ? wg : wb;
  int o = (idx < n4) ? idx : (idx - n4);
  float4 v = ((const float4*)src)[o];
  uint2 p;
  p.x = (unsigned)f2bf(v.x) | ((unsigned)f2bf(v.y) << 16);
  p.y = (unsigned)f2bf(v.z) | ((unsigned)f2bf(v.w) << 16);
  ((uint2*)out_bf)[idx] = p;                // gamma block then beta block
}

// ---------------------------------------------------------------------------
// Main fused kernel: 256 threads (8 wave32s), 32 rows per block.
// ---------------------------------------------------------------------------
__global__ __launch_bounds__(256) void adaln_main(
    const float* __restrict__ a, const float* __restrict__ s,
    const float* __restrict__ lnw, const float* __restrict__ bgam,
    const unsigned short* __restrict__ wgbf,
    const unsigned short* __restrict__ wbbf,
    float* __restrict__ out) {
  __shared__ unsigned short sh_sn[ROWS * SN_STRIDE];   // 25,088 B  (s_n bf16)
  __shared__ float          sh_an[ROWS * AN_STRIDE];   // 98,816 B  (a_n f32)
  __shared__ float          red_s[ROWS * 8];
  __shared__ float          red_q[ROWS * 8];

  const int tid     = threadIdx.x;
  const int rowbase = blockIdx.x * ROWS;

  // ---------------- Phase A1: LayerNorm(s) * lnw -> bf16 in LDS -----------
  {
    const int row = tid >> 3, j = tid & 7;             // 8 threads per row
    const float* srow = s + (size_t)(rowbase + row) * C_S;
    float4 v[12];
    float sum = 0.f, sq = 0.f;
#pragma unroll
    for (int i = 0; i < 12; ++i) {
      v[i] = ((const float4*)srow)[j + 8 * i];
      sum += v[i].x + v[i].y + v[i].z + v[i].w;
      sq  += v[i].x * v[i].x + v[i].y * v[i].y + v[i].z * v[i].z + v[i].w * v[i].w;
    }
    red_s[row * 8 + j] = sum;
    red_q[row * 8 + j] = sq;
    __syncthreads();
    sum = 0.f; sq = 0.f;
#pragma unroll
    for (int k = 0; k < 8; ++k) { sum += red_s[row * 8 + k]; sq += red_q[row * 8 + k]; }
    const float mu   = sum * (1.0f / C_S);
    const float var  = sq * (1.0f / C_S) - mu * mu;
    const float rstd = rsqrtf(var + LN_EPS);
#pragma unroll
    for (int i = 0; i < 12; ++i) {
      const int k4 = j + 8 * i;
      float4 w4 = ((const float4*)lnw)[k4];
      uint2 p;
      p.x = (unsigned)f2bf((v[i].x - mu) * rstd * w4.x) |
            ((unsigned)f2bf((v[i].y - mu) * rstd * w4.y) << 16);
      p.y = (unsigned)f2bf((v[i].z - mu) * rstd * w4.z) |
            ((unsigned)f2bf((v[i].w - mu) * rstd * w4.w) << 16);
      *(uint2*)&sh_sn[row * SN_STRIDE + k4 * 4] = p;
    }
  }
  __syncthreads();   // reduction buffers reused below

  // ---------------- Phase A2: LayerNorm(a) -> f32 in LDS ------------------
  {
    const int row = tid >> 3, j = tid & 7;
    const float* arow = a + (size_t)(rowbase + row) * C_A;
    float sum = 0.f, sq = 0.f;
#pragma unroll
    for (int i = 0; i < 24; ++i) {
      float4 v = ((const float4*)arow)[j + 8 * i];
      *(float4*)&sh_an[row * AN_STRIDE + (j + 8 * i) * 4] = v;   // stage raw a
      sum += v.x + v.y + v.z + v.w;
      sq  += v.x * v.x + v.y * v.y + v.z * v.z + v.w * v.w;
    }
    red_s[row * 8 + j] = sum;
    red_q[row * 8 + j] = sq;
    __syncthreads();
    sum = 0.f; sq = 0.f;
#pragma unroll
    for (int k = 0; k < 8; ++k) { sum += red_s[row * 8 + k]; sq += red_q[row * 8 + k]; }
    const float mu   = sum * (1.0f / C_A);
    const float var  = sq * (1.0f / C_A) - mu * mu;
    const float rstd = rsqrtf(var + LN_EPS);
#pragma unroll
    for (int i = 0; i < 24; ++i) {                     // normalize in place
      float4* p = (float4*)&sh_an[row * AN_STRIDE + (j + 8 * i) * 4];
      float4 v = *p;
      v.x = (v.x - mu) * rstd; v.y = (v.y - mu) * rstd;
      v.z = (v.z - mu) * rstd; v.w = (v.w - mu) * rstd;
      *p = v;
    }
  }
  __syncthreads();

  // ---------------- Phase B: dual GEMM (bf16 WMMA) + gated combine --------
  const int lane = tid & 31;
  const int wave = tid >> 5;
  const int nloc = lane & 15;      // column within 16-wide N tile / A-row
  const int lh   = lane >> 4;      // half select: K-half for frags, M-half for C
  const int rA0  = nloc;           // A-frag row, M-tile 0
  const int rA1  = 16 + nloc;      // A-frag row, M-tile 1

  for (int c = 0; c < 6; ++c) {    // 48 column tiles / 8 waves
    const int ct      = wave * 6 + c;
    const int colbase = ct * 16;
    const int col     = colbase + nloc;
    const unsigned short* bgp = wgbf + (size_t)col * C_S + lh * 8;
    const unsigned short* bbp = wbbf + (size_t)col * C_S + lh * 8;

    v8f accg0 = {0,0,0,0,0,0,0,0}, accg1 = {0,0,0,0,0,0,0,0};
    v8f accb0 = {0,0,0,0,0,0,0,0}, accb1 = {0,0,0,0,0,0,0,0};

#pragma unroll
    for (int kk = 0; kk < 12; ++kk) {                  // K = 384 = 12 * 32
      const int k0 = kk * 32;
      FragAB fa0, fa1, fg, fb;
      // A fragments from LDS (ISA 16-bit A layout: lanes<16 K{0..7,16..23},
      // lanes>=16 K{8..15,24..31})
      const unsigned short* a0p = &sh_sn[rA0 * SN_STRIDE + k0 + lh * 8];
      const unsigned short* a1p = &sh_sn[rA1 * SN_STRIDE + k0 + lh * 8];
      fa0.q[0] = *(const uint4*)(a0p);
      fa0.q[1] = *(const uint4*)(a0p + 16);
      fa1.q[0] = *(const uint4*)(a1p);
      fa1.q[1] = *(const uint4*)(a1p + 16);
      // B fragments from L2-resident bf16 weights (same K layout as A: K{8h..8h+7, 16+8h..23+8h})
      fg.q[0] = *(const uint4*)(bgp + k0);
      fg.q[1] = *(const uint4*)(bgp + k0 + 16);
      fb.q[0] = *(const uint4*)(bbp + k0);
      fb.q[1] = *(const uint4*)(bbp + k0 + 16);

      accg0 = __builtin_amdgcn_wmma_f32_16x16x32_bf16(false, fa0.v, false, fg.v,
                                                      (short)0, accg0, false, false);
      accg1 = __builtin_amdgcn_wmma_f32_16x16x32_bf16(false, fa1.v, false, fg.v,
                                                      (short)0, accg1, false, false);
      accb0 = __builtin_amdgcn_wmma_f32_16x16x32_bf16(false, fa0.v, false, fb.v,
                                                      (short)0, accb0, false, false);
      accb1 = __builtin_amdgcn_wmma_f32_16x16x32_bf16(false, fa1.v, false, fb.v,
                                                      (short)0, accb1, false, false);
    }

    // Epilogue: out = a_n * sigmoid(gamma + b_gamma) + beta
    const float bias = bgam[col];
#pragma unroll
    for (int mt = 0; mt < 2; ++mt) {
      const v8f g  = mt ? accg1 : accg0;
      const v8f bb = mt ? accb1 : accb0;
#pragma unroll
      for (int i = 0; i < 8; ++i) {
        const int r = mt * 16 + lh * 8 + i;            // C layout: VGPR i -> row lh*8+i
        const float an   = sh_an[r * AN_STRIDE + col];
        const float gate = 1.0f / (1.0f + __expf(-(g[i] + bias)));
        out[(size_t)(rowbase + r) * C_A + col] = an * gate + bb[i];
      }
    }
  }
}

// ---------------------------------------------------------------------------
extern "C" void kernel_launch(void* const* d_in, const int* in_sizes, int n_in,
                              void* d_out, int out_size, void* d_ws, size_t ws_size,
                              hipStream_t stream) {
  const float* a    = (const float*)d_in[0];   // [4,8192,768]
  const float* s    = (const float*)d_in[1];   // [4,8192,384]
  const float* lnw  = (const float*)d_in[2];   // [384]
  const float* wg   = (const float*)d_in[3];   // [768,384]
  const float* bgam = (const float*)d_in[4];   // [768]
  const float* wb   = (const float*)d_in[5];   // [768,384]
  float* out = (float*)d_out;                  // [4,8192,768]

  unsigned short* wgbf = (unsigned short*)d_ws;
  unsigned short* wbbf = wgbf + (size_t)C_A * C_S;

  // fp32 -> bf16 weights (1.18 MB, stays hot in the 192 MB L2)
  const int prep_threads = 2 * (C_A * C_S) / 4;        // 147456 float4 lanes
  adaln_prep_weights<<<(prep_threads + 255) / 256, 256, 0, stream>>>(wg, wb, wgbf);

  adaln_main<<<M_TOT / ROWS, 256, 0, stream>>>(a, s, lnw, bgam, wgbf, wbbf, out);
}
